// VAE_LSTM_ATTN_88923002896474
// MI455X (gfx1250) — matrix operand order repaired and checked by reading, not compiled, not run on hardware
//
#include <hip/hip_runtime.h>
#include <hip/hip_bf16.h>

typedef __bf16 bf16_t;
typedef __attribute__((ext_vector_type(16))) __bf16 bf16x16;
typedef __attribute__((ext_vector_type(8)))  __bf16 bf16x8;
typedef __attribute__((ext_vector_type(8)))  float  f32x8;

#define T_STEPS 2048
#define NWG_ENC 25

__device__ __forceinline__ float sigmoidf_(float v) {
    return 1.0f / (1.0f + __expf(-v));
}

__device__ __forceinline__ bf16x16 cat8(bf16x8 lo, bf16x8 hi) {
    return __builtin_shufflevector(lo, hi, 0,1,2,3,4,5,6,7,8,9,10,11,12,13,14,15);
}

// ---------------------------------------------------------------------------
// Kernel 0: pack weights into bf16 WMMA B-fragment layout, init h double-buffer
// B fragment layout assumption (lane l, element e): K = 32*kc + (l<16?0:16) + e,
//                                                   N = tile_base + (l & 15)
// ---------------------------------------------------------------------------
__global__ void prep_kernel(const float* __restrict__ x,
                            const float* __restrict__ W_enc_ih,   // [1600,4]
                            const float* __restrict__ W_enc_hh,   // [1600,400]
                            const float* __restrict__ W_dec_ih,   // [16,400]
                            bf16_t* __restrict__ WencP,           // [100*13*512]
                            bf16_t* __restrict__ WdecP,           // [13*512]
                            bf16_t* __restrict__ hbuf)            // [2][128][416]
{
    const size_t tid0   = (size_t)blockIdx.x * blockDim.x + threadIdx.x;
    const size_t stride = (size_t)gridDim.x * blockDim.x;

    // encoder combined weights: K<400 -> W_hh[n][k]; 400..403 -> W_ih[n][k-400]; else 0
    for (size_t p = tid0; p < 100ull * 13 * 512; p += stride) {
        int e    = (int)(p & 15);
        int lane = (int)((p >> 4) & 31);
        size_t frag = p >> 9;
        int kc = (int)(frag % 13);
        int nt = (int)(frag / 13);
        int N  = nt * 16 + (lane & 15);
        int K  = kc * 32 + ((lane >> 4) << 4) + e;
        float v = 0.0f;
        if (K < 400)       v = W_enc_hh[(size_t)N * 400 + K];
        else if (K < 404)  v = W_enc_ih[(size_t)N * 4 + (K - 400)];
        WencP[p] = (bf16_t)v;
    }
    // decoder input-to-gate weights (one 16-wide N tile)
    for (size_t p = tid0; p < 13ull * 512; p += stride) {
        int e    = (int)(p & 15);
        int lane = (int)((p >> 4) & 31);
        int kc   = (int)(p >> 9);
        int N    = lane & 15;
        int K    = kc * 32 + ((lane >> 4) << 4) + e;
        float v  = (K < 400) ? W_dec_ih[(size_t)N * 400 + K] : 0.0f;
        WdecP[p] = (bf16_t)v;
    }
    // h double buffer: zeros; buf0 cols 400..403 = x[:,0,:]
    for (size_t p = tid0; p < 2ull * 128 * 416; p += stride) {
        size_t q = p % (128 * 416);
        int b   = (int)(q / 416);
        int col = (int)(q % 416);
        float v = 0.0f;
        if (p < (size_t)128 * 416 && col >= 400 && col < 404)
            v = x[((size_t)b * T_STEPS + 0) * 4 + (col - 400)];
        hbuf[p] = (bf16_t)v;
    }
}

// ---------------------------------------------------------------------------
// Kernel 1: persistent encoder LSTM. 25 WGs x 256 threads (8 waves).
// WG owns 16 h-columns; wave owns one 16-row batch tile and all 4 gates of it.
// c-state stays in registers; grid barrier per timestep via agent-scope atomics.
// ---------------------------------------------------------------------------
__global__ __launch_bounds__(256) void enc_kernel(const float* __restrict__ x,
                                                  const float* __restrict__ b_enc,
                                                  const bf16_t* __restrict__ WencP,
                                                  bf16_t* __restrict__ hbuf,
                                                  unsigned int* __restrict__ ctr)
{
    __shared__ __align__(32) bf16_t ldsB[4 * 13 * 512];   // 52 KB

    const int tid = threadIdx.x;
    const int wg  = blockIdx.x;          // 0..24
    const int j0  = wg * 16;             // first h-column owned

    // stage this WG's B fragments (4 gates x 13 K-chunks) into LDS
    {
        unsigned int* dst = (unsigned int*)ldsB;
        for (int i = tid; i < 4 * 13 * 256; i += 256) {    // 256 dwords / fragment
            int frag = i >> 8;
            int off  = i & 255;
            int G  = frag / 13;
            int kc = frag % 13;
            const unsigned int* src =
                (const unsigned int*)(WencP + ((size_t)((G * 25 + wg) * 13 + kc)) * 512);
            dst[frag * 256 + off] = src[off];
        }
    }
    __syncthreads();

    const int wave  = tid >> 5;
    const int lane  = tid & 31;
    const int m     = wave;                    // M tile (batch rows 16m..16m+15)
    const int laneN = lane & 15;
    const int half  = lane >> 4;               // 0/1
    const int arow  = m * 16 + laneN;          // A-fragment source row

    const float bias_i = b_enc[0 * 400 + j0 + laneN];
    const float bias_f = b_enc[1 * 400 + j0 + laneN];
    const float bias_g = b_enc[2 * 400 + j0 + laneN];
    const float bias_o = b_enc[3 * 400 + j0 + laneN];

    f32x8 cst = {0.f, 0.f, 0.f, 0.f, 0.f, 0.f, 0.f, 0.f};

    for (int t = 0; t < T_STEPS; ++t) {
        const bf16_t* hr = hbuf + (size_t)(t & 1) * 128 * 416;
        bf16_t*       hw = hbuf + (size_t)((t + 1) & 1) * 128 * 416;

        f32x8 ai = {bias_i, bias_i, bias_i, bias_i, bias_i, bias_i, bias_i, bias_i};
        f32x8 af = {bias_f, bias_f, bias_f, bias_f, bias_f, bias_f, bias_f, bias_f};
        f32x8 ag = {bias_g, bias_g, bias_g, bias_g, bias_g, bias_g, bias_g, bias_g};
        f32x8 ao = {bias_o, bias_o, bias_o, bias_o, bias_o, bias_o, bias_o, bias_o};

        #pragma unroll
        for (int kc = 0; kc < 13; ++kc) {
            const bf16_t* ap = hr + (size_t)arow * 416 + kc * 32 + half * 16;
            bf16x8 alo = *(const bf16x8*)ap;
            bf16x8 ahi = *(const bf16x8*)(ap + 8);
            bf16x16 a  = cat8(alo, ahi);

            bf16x16 b0 = *(const bf16x16*)(ldsB + ((0 * 13 + kc) * 512 + lane * 16));
            bf16x16 b1 = *(const bf16x16*)(ldsB + ((1 * 13 + kc) * 512 + lane * 16));
            bf16x16 b2 = *(const bf16x16*)(ldsB + ((2 * 13 + kc) * 512 + lane * 16));
            bf16x16 b3 = *(const bf16x16*)(ldsB + ((3 * 13 + kc) * 512 + lane * 16));

            ai = __builtin_amdgcn_wmma_f32_16x16x32_bf16(false, a, false, b0, (short)0, ai, false, false);
            af = __builtin_amdgcn_wmma_f32_16x16x32_bf16(false, a, false, b1, (short)0, af, false, false);
            ag = __builtin_amdgcn_wmma_f32_16x16x32_bf16(false, a, false, b2, (short)0, ag, false, false);
            ao = __builtin_amdgcn_wmma_f32_16x16x32_bf16(false, a, false, b3, (short)0, ao, false, false);
        }

        // gate nonlinearities + state update, write h slice (bf16) to next buffer
        #pragma unroll
        for (int r = 0; r < 8; ++r) {
            float iv = sigmoidf_(ai[r]);
            float fv = sigmoidf_(af[r]);
            float gv = tanhf(ag[r]);
            float ov = sigmoidf_(ao[r]);
            float c  = fv * cst[r] + iv * gv;
            cst[r]   = c;
            float h  = ov * tanhf(c);
            int row  = m * 16 + r + half * 8;
            hw[(size_t)row * 416 + j0 + laneN] = (bf16_t)h;
        }

        // WG0 refreshes the x_t columns of the write buffer for step t+1
        if (wg == 0 && (t + 1) < T_STEPS) {
            for (int e = tid; e < 512; e += 256) {
                int b  = e >> 2;
                int ii = e & 3;
                hw[(size_t)b * 416 + 400 + ii] =
                    (bf16_t)x[((size_t)b * T_STEPS + (t + 1)) * 4 + ii];
            }
        }

        // grid-wide barrier (device scope)
        __threadfence();
        __syncthreads();
        if (tid == 0) {
            __hip_atomic_fetch_add(&ctr[t], 1u, __ATOMIC_ACQ_REL, __HIP_MEMORY_SCOPE_AGENT);
            while (__hip_atomic_load(&ctr[t], __ATOMIC_ACQUIRE, __HIP_MEMORY_SCOPE_AGENT) < NWG_ENC) {
                __builtin_amdgcn_s_sleep(1);
            }
        }
        __syncthreads();
    }
}

// ---------------------------------------------------------------------------
// Kernel 2: mu / logvar / z / hd0 (tiny)
// ---------------------------------------------------------------------------
__global__ __launch_bounds__(256) void latent_kernel(const bf16_t* __restrict__ hbuf, // buf0 = final h
                                                     const float* __restrict__ eps,
                                                     const float* __restrict__ W_mu,
                                                     const float* __restrict__ b_mu,
                                                     const float* __restrict__ W_lv,
                                                     const float* __restrict__ b_lv,
                                                     const float* __restrict__ W_l2h,
                                                     const float* __restrict__ b_l2h,
                                                     float* __restrict__ out_mu,
                                                     float* __restrict__ out_lv,
                                                     float* __restrict__ hd0)
{
    __shared__ float zbuf[128 * 20];
    const int tid = threadIdx.x;
    for (int o = tid; o < 128 * 20; o += 256) {
        int b = o / 20, l = o % 20;
        const bf16_t* hr = hbuf + (size_t)b * 416;
        float sm = 0.f, sv = 0.f;
        for (int k = 0; k < 400; ++k) {
            float h = (float)hr[k];
            sm += h * W_mu[(size_t)l * 400 + k];
            sv += h * W_lv[(size_t)l * 400 + k];
        }
        sm += b_mu[l];
        sv += b_lv[l];
        out_mu[o] = sm;
        out_lv[o] = sv;
        zbuf[o] = sm + eps[o] * __expf(0.5f * sv);
    }
    __syncthreads();
    for (int o = tid; o < 512; o += 256) {
        int b = o >> 2, i = o & 3;
        float s = b_l2h[i];
        for (int l = 0; l < 20; ++l) s += zbuf[b * 20 + l] * W_l2h[i * 20 + l];
        hd0[o] = s;
    }
}

// ---------------------------------------------------------------------------
// Kernel 3: uw[t][b][g] = relu(x_t @ W_i2h^T + b_i2h) @ W_dec_ih^T + b_dec
// One block per t; K staged in 32-wide LDS chunks; 13 WMMAs per wave.
// ---------------------------------------------------------------------------
__global__ __launch_bounds__(256) void uw_kernel(const float* __restrict__ x,
                                                 const float* __restrict__ W_i2h,   // [400,4]
                                                 const float* __restrict__ b_i2h,  // [400]
                                                 const bf16_t* __restrict__ WdecP, // [13*512]
                                                 const float* __restrict__ b_dec,  // [16]
                                                 float* __restrict__ uw)           // [2048][128][16]
{
    __shared__ __align__(32) bf16_t ldsU[128 * 32];    // 8 KB
    const int t    = blockIdx.x;
    const int tid  = threadIdx.x;
    const int wave = tid >> 5;
    const int lane = tid & 31;
    const int laneN = lane & 15;
    const int half  = lane >> 4;
    const int arow  = wave * 16 + laneN;

    const float bias = b_dec[laneN];
    f32x8 acc = {bias, bias, bias, bias, bias, bias, bias, bias};

    for (int kc = 0; kc < 13; ++kc) {
        // stage u[:, 32*kc .. 32*kc+31] (relu of input projection) as bf16
        for (int p = tid; p < 128 * 32; p += 256) {
            int b  = p >> 5;
            int kl = p & 31;
            int K  = kc * 32 + kl;
            float v = 0.0f;
            if (K < 400) {
                const float* xr = x + ((size_t)b * T_STEPS + t) * 4;
                const float* wr = W_i2h + (size_t)K * 4;
                v = xr[0] * wr[0] + xr[1] * wr[1] + xr[2] * wr[2] + xr[3] * wr[3] + b_i2h[K];
                v = fmaxf(v, 0.0f);
            }
            ldsU[p] = (bf16_t)v;
        }
        __syncthreads();

        const bf16_t* ap = ldsU + (size_t)arow * 32 + half * 16;
        bf16x8 alo = *(const bf16x8*)ap;
        bf16x8 ahi = *(const bf16x8*)(ap + 8);
        bf16x16 a  = cat8(alo, ahi);
        bf16x16 bf = *(const bf16x16*)(WdecP + (size_t)kc * 512 + lane * 16);
        acc = __builtin_amdgcn_wmma_f32_16x16x32_bf16(false, a, false, bf, (short)0, acc, false, false);
        __syncthreads();
    }

    #pragma unroll
    for (int r = 0; r < 8; ++r) {
        int row = wave * 16 + r + half * 8;
        uw[((size_t)t * 128 + row) * 16 + laneN] = acc[r];
    }
}

// ---------------------------------------------------------------------------
// Kernel 4: decoder LSTM scan (hidden size 4), one thread per batch row
// ---------------------------------------------------------------------------
__global__ __launch_bounds__(128) void dec_kernel(const float* __restrict__ uw,
                                                  const float* __restrict__ hd0,
                                                  const float* __restrict__ W_dec_hh, // [16,4]
                                                  float* __restrict__ out_sample)     // [128,4]
{
    __shared__ float sW[64];
    const int tid = threadIdx.x;   // batch index
    if (tid < 64) sW[tid] = W_dec_hh[tid];
    __syncthreads();

    float h[4], c[4] = {0.f, 0.f, 0.f, 0.f};
    #pragma unroll
    for (int k = 0; k < 4; ++k) h[k] = hd0[tid * 4 + k];

    for (int t = 0; t < T_STEPS; ++t) {
        const float* g = uw + ((size_t)t * 128 + tid) * 16;
        float gg[16];
        #pragma unroll
        for (int q = 0; q < 16; ++q)
            gg[q] = g[q] + h[0] * sW[q * 4 + 0] + h[1] * sW[q * 4 + 1]
                         + h[2] * sW[q * 4 + 2] + h[3] * sW[q * 4 + 3];
        #pragma unroll
        for (int k = 0; k < 4; ++k) {
            float iv = sigmoidf_(gg[k]);
            float fv = sigmoidf_(gg[4 + k]);
            float gv = tanhf(gg[8 + k]);
            float ov = sigmoidf_(gg[12 + k]);
            c[k] = fv * c[k] + iv * gv;
            h[k] = ov * tanhf(c[k]);
        }
    }
    #pragma unroll
    for (int k = 0; k < 4; ++k) out_sample[tid * 4 + k] = h[k];
}

// ---------------------------------------------------------------------------
extern "C" void kernel_launch(void* const* d_in, const int* in_sizes, int n_in,
                              void* d_out, int out_size, void* d_ws, size_t ws_size,
                              hipStream_t stream) {
    const float* x        = (const float*)d_in[0];
    const float* eps      = (const float*)d_in[1];
    const float* W_enc_ih = (const float*)d_in[2];
    const float* W_enc_hh = (const float*)d_in[3];
    const float* b_enc    = (const float*)d_in[4];
    const float* W_mu     = (const float*)d_in[5];
    const float* b_mu     = (const float*)d_in[6];
    const float* W_lv     = (const float*)d_in[7];
    const float* b_lv     = (const float*)d_in[8];
    const float* W_l2h    = (const float*)d_in[9];
    const float* b_l2h    = (const float*)d_in[10];
    const float* W_i2h    = (const float*)d_in[11];
    const float* b_i2h    = (const float*)d_in[12];
    const float* W_dec_ih = (const float*)d_in[13];
    const float* W_dec_hh = (const float*)d_in[14];
    const float* b_dec    = (const float*)d_in[15];

    float* out = (float*)d_out;   // [sample 512 | mu 2560 | logvar 2560]

    char* ws = (char*)d_ws;
    size_t off = 0;
    auto take = [&](size_t bytes) { size_t o = off; off = (off + bytes + 255) & ~(size_t)255; return o; };
    unsigned int* ctr = (unsigned int*)(ws + take((size_t)T_STEPS * 4));
    bf16_t* WencP     = (bf16_t*)(ws + take(100ull * 13 * 512 * 2));
    bf16_t* WdecP     = (bf16_t*)(ws + take(13ull * 512 * 2));
    bf16_t* hbuf      = (bf16_t*)(ws + take(2ull * 128 * 416 * 2));
    float*  hd0       = (float*)(ws + take(512ull * 4));
    float*  uw        = (float*)(ws + take((size_t)T_STEPS * 128 * 16 * 4));

    hipMemsetAsync(ctr, 0, (size_t)T_STEPS * 4, stream);
    prep_kernel<<<512, 256, 0, stream>>>(x, W_enc_ih, W_enc_hh, W_dec_ih, WencP, WdecP, hbuf);
    enc_kernel<<<NWG_ENC, 256, 0, stream>>>(x, b_enc, WencP, hbuf, ctr);
    latent_kernel<<<1, 256, 0, stream>>>(hbuf, eps, W_mu, b_mu, W_lv, b_lv, W_l2h, b_l2h,
                                         out + 512, out + 512 + 2560, hd0);
    uw_kernel<<<T_STEPS, 256, 0, stream>>>(x, W_i2h, b_i2h, WdecP, b_dec, uw);
    dec_kernel<<<1, 128, 0, stream>>>(uw, hd0, W_dec_hh, out);
}
